// ValueNetwork__9706626089212
// MI455X (gfx1250) — matrix operand order repaired and checked by reading, not compiled, not run on hardware
//
#include <hip/hip_runtime.h>
#include <hip/hip_bf16.h>

typedef __attribute__((ext_vector_type(16))) _Float16 v16h;
typedef __attribute__((ext_vector_type(8)))  float    v8f;
typedef __attribute__((ext_vector_type(4)))  float    v4f;
typedef __attribute__((ext_vector_type(2)))  float    v2f;

#define BATCH   16384
#define NAG     8
#define BN      (BATCH * NAG)      // 131072 rows
#define SDIM    128
#define ADIM    6
#define MDIM    7                  // N-1
#define ROWS    64                 // rows per block
#define KV      192                // value-input K padded (170 -> 192)

// workspace layout (in _Float16 elements)
#define W1H_OFF 0                       // 512 x 128
#define W2H_OFF 65536                   // 256 x 512
#define WFH_OFF 196608                  // 16  x 256  (rows 14..15 zero)
#define V1H_OFF 200704                  // 512 x 192  (cols 170..191 zero)
#define V2H_OFF 299008                  // 256 x 512

// LDS layout (bytes)
#define SXH_OFF 0                       // 64 x 128 f16   = 16384
#define SH1_OFF 16384                   // 64 x 512 f16   = 65536
#define SH2_OFF 81920                   // 64 x 256 f16   = 32768
#define SL3_OFF 114688                  // 64 x 16  f32   = 4096
#define SSV_OFF 118784                  // 64 x 192 f16   = 24576
#define SMEM_BYTES 143360

// ---------------------------------------------------------------------------
// fp32 -> f16 weight conversion with zero padding
// ---------------------------------------------------------------------------
__global__ void cvt_pad_f16(const float* __restrict__ src, _Float16* __restrict__ dst,
                            int sR, int sC, int dR, int dC) {
    int i = blockIdx.x * 256 + threadIdx.x;
    if (i >= dR * dC) return;
    int r = i / dC, c = i - r * dC;
    float v = (r < sR && c < sC) ? src[r * sC + c] : 0.0f;
    dst[i] = (_Float16)v;
}

// ---------------------------------------------------------------------------
// Register-blocked WMMA: 2 row tiles x 4 adjacent column tiles per wave.
// Per K-step: 2 A fragments (4 ds_load_b128, reused 4x each) + 4 B fragments
// (8 global_load_b128, reused 2x each, loaded up-front so they clause and
// their latency overlaps the 8 WMMAs).
//
// A layout per ISA (16-bit A 16x32): lanes 0-15 row M, K chunks at 0..7/16..23;
// lanes 16-31 row M, K chunks at 8..15/24..31 -> two 16B LDS loads.
// B layout (32x16): lane = column (lane&15), K chunks same as A ->
// two 16B global loads per column tile (weights row-major (out, K)).
// ---------------------------------------------------------------------------
template <int K, int LDA>
__device__ __forceinline__ void tile_mm_2x4(const _Float16* sA, const _Float16* gB,
                                            int m0, int n0, int lane, v8f c[2][4]) {
    const int sel = lane >> 4;
    const int mr  = lane & 15;
    const _Float16* aRow0 = sA + (m0 + mr) * LDA;
    const _Float16* aRow1 = aRow0 + 16 * LDA;
#pragma unroll
    for (int kb = 0; kb < K; kb += 32) {
        union { v16h v; v4f f[2]; } a0, a1, b[4];
        // B fragments first: 8 global loads clause together, latency hidden
        // behind the WMMA burst below.
#pragma unroll
        for (int j = 0; j < 4; ++j) {
            const _Float16* bRow = gB + (size_t)(n0 + j * 16 + mr) * K;
            b[j].f[0] = *(const v4f*)(bRow + kb + sel * 8);
            b[j].f[1] = *(const v4f*)(bRow + kb + 16 + sel * 8);
        }
        a0.f[0] = *(const v4f*)(aRow0 + kb + sel * 8);
        a0.f[1] = *(const v4f*)(aRow0 + kb + 16 + sel * 8);
        a1.f[0] = *(const v4f*)(aRow1 + kb + sel * 8);
        a1.f[1] = *(const v4f*)(aRow1 + kb + 16 + sel * 8);
#pragma unroll
        for (int j = 0; j < 4; ++j) {
            c[0][j] = __builtin_amdgcn_wmma_f32_16x16x32_f16(false, a0.v, false, b[j].v,
                                                             (short)0, c[0][j], false, false);
            c[1][j] = __builtin_amdgcn_wmma_f32_16x16x32_f16(false, a1.v, false, b[j].v,
                                                             (short)0, c[1][j], false, false);
        }
    }
}

// single-tile variant (used for the tiny 14-wide head)
template <int K, int LDA>
__device__ __forceinline__ v8f tile_mm(const _Float16* sA, const _Float16* gB,
                                       int m0, int n0, int lane) {
    v8f c = {};
    const int sel = lane >> 4;
    const int mr  = lane & 15;
    const _Float16* aRow = sA + (m0 + mr) * LDA;
    const _Float16* bRow = gB + (size_t)(n0 + mr) * K;
#pragma unroll
    for (int kb = 0; kb < K; kb += 32) {
        union { v16h v; v4f f[2]; } a, b;
        a.f[0] = *(const v4f*)(aRow + kb + sel * 8);
        a.f[1] = *(const v4f*)(aRow + kb + 16 + sel * 8);
        b.f[0] = *(const v4f*)(bRow + kb + sel * 8);
        b.f[1] = *(const v4f*)(bRow + kb + 16 + sel * 8);
        c = __builtin_amdgcn_wmma_f32_16x16x32_f16(false, a.v, false, b.v,
                                                   (short)0, c, false, false);
    }
    return c;
}

// store D (+bias, relu) as f16 into LDS dest (row-major, stride ldd)
__device__ __forceinline__ void store_relu_f16(v8f c, _Float16* sD, int ldd,
                                               int m0, int n0, int lane, float bias) {
    const int col = n0 + (lane & 15);
    const int sel = lane >> 4;
#pragma unroll
    for (int r = 0; r < 8; ++r) {
        float v = c[r] + bias;
        v = v > 0.0f ? v : 0.0f;
        sD[(m0 + r + sel * 8) * ldd + col] = (_Float16)v;
    }
}

// ---------------------------------------------------------------------------
// Fused policy + mixing + value kernel. One block = 64 rows, 8 wave32.
// Wave w owns row tiles {rp*32, rp*32+16} (rp = w&1) and column quarter
// cs = w>>1, so each weight element is fetched only twice per block.
// ---------------------------------------------------------------------------
__global__ __launch_bounds__(256) void fused_ctde_kernel(
    const float* __restrict__ states,       // (B, 8, 128)
    const float* __restrict__ one_hot,      // (B, 8, 6)
    const float* __restrict__ probs,        // (B, 8, 6)
    const float* __restrict__ b1,           // (512)
    const float* __restrict__ b2,           // (256)
    const float* __restrict__ bf,           // (14)
    const float* __restrict__ vb1,          // (512)
    const float* __restrict__ vb2,          // (256)
    const float* __restrict__ Vo,           // (1, 256)
    const float* __restrict__ vbo,          // (1)
    const _Float16* __restrict__ wsh,       // converted f16 weights
    float* __restrict__ out)                // wa | wp | value
{
    extern __shared__ char smem[];
    _Float16* sXh = (_Float16*)(smem + SXH_OFF);   // 64 x 128
    _Float16* sH1 = (_Float16*)(smem + SH1_OFF);   // 64 x 512 (reused for HV1)
    _Float16* sH2 = (_Float16*)(smem + SH2_OFF);   // 64 x 256 (reused for HV2)
    float*    sL3 = (float*)   (smem + SL3_OFF);   // 64 x 16
    _Float16* sSV = (_Float16*)(smem + SSV_OFF);   // 64 x 192

    const _Float16* W1h = wsh + W1H_OFF;
    const _Float16* W2h = wsh + W2H_OFF;
    const _Float16* Wfh = wsh + WFH_OFF;
    const _Float16* V1h = wsh + V1H_OFF;
    const _Float16* V2h = wsh + V2H_OFF;

    const int tid  = threadIdx.x;
    const int w    = tid >> 5;
    const int lane = tid & 31;
    const int row0 = blockIdx.x * ROWS;

    const int rp = w & 1;         // row pair: tiles rp*32 and rp*32+16
    const int cs = w >> 1;        // column quarter
    const int m0 = rp * 32;

    // warm L2/WGP$ for the weight streams (global_prefetch_b8)
    __builtin_prefetch(W1h + tid * 256, 0, 1);
    __builtin_prefetch(W2h + tid * 512, 0, 1);
    __builtin_prefetch(V1h + tid * 384, 0, 1);
    __builtin_prefetch(V2h + tid * 512, 0, 1);

    // ---- stage 0: states tile -> LDS f16 (vectorized: float4 -> 4xf16) --
    for (int i = tid; i < ROWS * SDIM / 4; i += 256) {
        v4f x = ((const v4f*)states)[(size_t)row0 * (SDIM / 4) + i];
        union { _Float16 h[4]; v2f f; } u;
        u.h[0] = (_Float16)x[0]; u.h[1] = (_Float16)x[1];
        u.h[2] = (_Float16)x[2]; u.h[3] = (_Float16)x[3];
        *(v2f*)(sXh + i * 4) = u.f;
    }
    __syncthreads();

    // ---- L1: (64x128) @ W1^T -> relu -> sH1 (64x512) --------------------
#pragma unroll
    for (int g = 0; g < 2; ++g) {
        int n0 = cs * 64 + g * 256;     // this wave's 64-col group (of 512)
        v8f c[2][4] = {};
        tile_mm_2x4<128, 128>(sXh, W1h, m0, n0, lane, c);
#pragma unroll
        for (int p = 0; p < 2; ++p)
#pragma unroll
            for (int j = 0; j < 4; ++j)
                store_relu_f16(c[p][j], sH1, 512, m0 + p * 16, n0 + j * 16, lane,
                               b1[n0 + j * 16 + (lane & 15)]);
    }
    __syncthreads();

    // ---- L2: (64x512) @ W2^T -> relu -> sH2 (64x256) --------------------
    {
        int n0 = cs * 32;               // 32 cols of 256... need 4 tiles -> 64
        // 256 cols / 8 waves = 32 cols = 2 tiles; use 2x4 over 64 cols for
        // even waves' group and odd waves handle the other row pair of same
        // columns -> instead: each (rp, cs) pair owns 4 tiles at cs*32..+63
        // only when cs < 4 would overlap. Use explicit mapping: 4 col groups
        // of 64, each handled by two waves (rp 0/1).
        n0 = cs * 32;                   // kept simple: 8 waves x 2 tiles
        v8f c[2][4] = {};
        // 2 real column tiles per wave; compute a 4-group at aligned base and
        // keep all (cheap, uniform code path): base = (cs & ~1) * 32
        int base = (cs >> 1) * 64;
        tile_mm_2x4<512, 512>(sH1, W2h, (cs & 1) * 32, base, lane, c);
#pragma unroll
        for (int p = 0; p < 2; ++p)
#pragma unroll
            for (int j = 0; j < 4; ++j)
                store_relu_f16(c[p][j], sH2, 256, (cs & 1) * 32 + p * 16, base + j * 16,
                               lane, b2[base + j * 16 + (lane & 15)]);
    }
    __syncthreads();

    // ---- L3: (64x256) @ Wf^T (padded 16x256) -> sL3 f32 (64x16) ---------
    if (w < 4) {
        v8f c = tile_mm<256, 256>(sH2, Wfh, w * 16, 0, lane);
        const int col = lane & 15, sel = lane >> 4;
#pragma unroll
        for (int r = 0; r < 8; ++r)
            sL3[(w * 16 + r + sel * 8) * 16 + col] = c[r];
    }
    __syncthreads();

    // ---- per-row: bias + pairwise softmax, outputs, build value input ---
    if (tid < ROWS) {
        const int rowg = row0 + tid;
        const int bidx = rowg >> 3;
        const int nag  = rowg & 7;

        float wa[MDIM], wp[MDIM];
#pragma unroll
        for (int j = 0; j < MDIM; ++j) {
            float p0 = sL3[tid * 16 + 2 * j]     + bf[2 * j];
            float p1 = sL3[tid * 16 + 2 * j + 1] + bf[2 * j + 1];
            float mx = p0 > p1 ? p0 : p1;
            float e0 = __expf(p0 - mx);
            float e1 = __expf(p1 - mx);
            float inv = 1.0f / (e0 + e1);
            wp[j] = e0 * inv;   // weight_prob
            wa[j] = e1 * inv;   // weight_action
        }
        float* out_wa = out;
        float* out_wp = out + (size_t)BN * MDIM;
#pragma unroll
        for (int j = 0; j < MDIM; ++j) {
            out_wa[(size_t)rowg * MDIM + j] = wa[j];
            out_wp[(size_t)rowg * MDIM + j] = wp[j];
        }

        // z[j][k] = wa[j]*one_hot[loo] + wp[j]*probs[loo]
        float z[MDIM][ADIM];
#pragma unroll
        for (int j = 0; j < MDIM; ++j) {
            int idx = j + (j >= nag ? 1 : 0);
            const float* oh = one_hot + ((size_t)bidx * NAG + idx) * ADIM;
            const float* pr = probs   + ((size_t)bidx * NAG + idx) * ADIM;
#pragma unroll
            for (int k = 0; k < ADIM; ++k)
                z[j][k] = wa[j] * oh[k] + wp[j] * pr[k];
        }

        // states_value = [states[:92] | (z[j], chunk[j]) x6 | z[6]] padded to 192
        _Float16* sv = sSV + tid * KV;
        const _Float16* xr = sXh + tid * SDIM;
        for (int c = 0; c < 92; ++c) sv[c] = xr[c];
        int pos = 92;
#pragma unroll
        for (int j = 0; j < MDIM - 1; ++j) {
#pragma unroll
            for (int k = 0; k < ADIM; ++k) sv[pos + k] = (_Float16)z[j][k];
#pragma unroll
            for (int k = 0; k < 6; ++k)    sv[pos + 6 + k] = xr[92 + j * 6 + k];
            pos += 12;
        }
#pragma unroll
        for (int k = 0; k < ADIM; ++k) sv[164 + k] = (_Float16)z[MDIM - 1][k];
        for (int c = 170; c < KV; ++c) sv[c] = (_Float16)0.0f;
    }
    __syncthreads();

    // ---- L4: (64x192) @ V1^T (padded 512x192) -> relu -> sH1 (64x512) ---
#pragma unroll
    for (int g = 0; g < 2; ++g) {
        int n0 = cs * 64 + g * 256;
        v8f c[2][4] = {};
        tile_mm_2x4<KV, KV>(sSV, V1h, m0, n0, lane, c);
#pragma unroll
        for (int p = 0; p < 2; ++p)
#pragma unroll
            for (int j = 0; j < 4; ++j)
                store_relu_f16(c[p][j], sH1, 512, m0 + p * 16, n0 + j * 16, lane,
                               vb1[n0 + j * 16 + (lane & 15)]);
    }
    __syncthreads();

    // ---- L5: (64x512) @ V2^T -> relu -> sH2 (64x256) --------------------
    {
        int base = (cs >> 1) * 64;
        v8f c[2][4] = {};
        tile_mm_2x4<512, 512>(sH1, V2h, (cs & 1) * 32, base, lane, c);
#pragma unroll
        for (int p = 0; p < 2; ++p)
#pragma unroll
            for (int j = 0; j < 4; ++j)
                store_relu_f16(c[p][j], sH2, 256, (cs & 1) * 32 + p * 16, base + j * 16,
                               lane, vb2[base + j * 16 + (lane & 15)]);
    }
    __syncthreads();

    // ---- L6: (64x256) @ Vo^T (256 -> 1), scalar dot per row -------------
    if (tid < ROWS) {
        const int rowg = row0 + tid;
        float acc = vbo[0];
        const _Float16* hr = sH2 + tid * 256;
        for (int k = 0; k < 256; ++k)
            acc += (float)hr[k] * Vo[k];
        out[(size_t)2 * BN * MDIM + rowg] = acc;
    }
}

// ---------------------------------------------------------------------------
extern "C" void kernel_launch(void* const* d_in, const int* in_sizes, int n_in,
                              void* d_out, int out_size, void* d_ws, size_t ws_size,
                              hipStream_t stream) {
    const float* states  = (const float*)d_in[0];
    const float* one_hot = (const float*)d_in[1];
    const float* probs   = (const float*)d_in[2];
    const float* W1      = (const float*)d_in[3];
    const float* b1      = (const float*)d_in[4];
    const float* W2      = (const float*)d_in[5];
    const float* b2      = (const float*)d_in[6];
    const float* Wf      = (const float*)d_in[7];
    const float* bf      = (const float*)d_in[8];
    const float* V1      = (const float*)d_in[9];
    const float* vb1     = (const float*)d_in[10];
    const float* V2      = (const float*)d_in[11];
    const float* vb2     = (const float*)d_in[12];
    const float* Vo      = (const float*)d_in[13];
    const float* vbo     = (const float*)d_in[14];

    _Float16* wsh = (_Float16*)d_ws;

    // convert weights to f16 WMMA-friendly (row-major (out, K), zero-padded)
    auto blocks = [](int n) { return (n + 255) / 256; };
    cvt_pad_f16<<<blocks(512 * 128), 256, 0, stream>>>(W1, wsh + W1H_OFF, 512, 128, 512, 128);
    cvt_pad_f16<<<blocks(256 * 512), 256, 0, stream>>>(W2, wsh + W2H_OFF, 256, 512, 256, 512);
    cvt_pad_f16<<<blocks(16  * 256), 256, 0, stream>>>(Wf, wsh + WFH_OFF, 14, 256, 16, 256);
    cvt_pad_f16<<<blocks(512 * KV ), 256, 0, stream>>>(V1, wsh + V1H_OFF, 512, 170, 512, KV);
    cvt_pad_f16<<<blocks(256 * 512), 256, 0, stream>>>(V2, wsh + V2H_OFF, 256, 512, 256, 512);

    dim3 grid(BN / ROWS);   // 2048 blocks of 64 rows
    dim3 block(256);        // 8 wave32
    fused_ctde_kernel<<<grid, block, SMEM_BYTES, stream>>>(
        states, one_hot, probs, b1, b2, bf, vb1, vb2, Vo, vbo, wsh, (float*)d_out);
}
